// CLIPAttention_25718264168860
// MI455X (gfx1250) — matrix operand order repaired and checked by reading, not compiled, not run on hardware
//
#include <hip/hip_runtime.h>
#include <hip/hip_bf16.h>

typedef __bf16 bf16_t;
typedef __attribute__((ext_vector_type(16))) __bf16 bf16x16;
typedef __attribute__((ext_vector_type(8)))  __bf16 bf16x8;
typedef __attribute__((ext_vector_type(8)))  float  f32x8;

#define EMBED 768
#define NHEAD 12
#define HDIM  64
#define SEQ   1024
#define NBATCH 8
#define QK_SCALE 0.125f   // 64^-0.5

// ---------------------------------------------------------------- async copy
// CDNA5 global->LDS async DMA (ASYNCcnt), GV mode: vdst = wave-relative LDS
// byte offset (== low 32 bits of the generic shared pointer per the aperture
// rule), vaddr = 64-bit global address. 16 bytes per lane per op.
__device__ __forceinline__ void async_copy16(unsigned int lds_off, const void* g) {
    asm volatile("global_load_async_to_lds_b128 %0, %1, off"
                 :: "v"(lds_off), "v"((unsigned long long)(uintptr_t)g)
                 : "memory");
}
__device__ __forceinline__ void wait_async0() {
    asm volatile("s_wait_asynccnt 0x0" ::: "memory");
}

// ---------------------------------------------------------------- prep kernels

__global__ __launch_bounds__(256)
void cvt_f32_bf16(const float* __restrict__ in, bf16_t* __restrict__ out, int n4) {
    int i = blockIdx.x * 256 + threadIdx.x;
    if (i < n4) {
        float4 v = ((const float4*)in)[i];
        bf16_t* o = out + i * 4;
        o[0] = (bf16_t)v.x; o[1] = (bf16_t)v.y; o[2] = (bf16_t)v.z; o[3] = (bf16_t)v.w;
    }
}

// W [k][n] fp32 -> WT [n][k] bf16   (E x E)
__global__ __launch_bounds__(256)
void transpose_w_bf16(const float* __restrict__ W, bf16_t* __restrict__ WT) {
    __shared__ float tile[32][33];
    int n0 = blockIdx.x * 32, k0 = blockIdx.y * 32;
    int tx = threadIdx.x, ty = threadIdx.y;          // 32 x 8
    #pragma unroll
    for (int i = 0; i < 32; i += 8)
        tile[ty + i][tx] = W[(size_t)(k0 + ty + i) * EMBED + n0 + tx];
    __syncthreads();
    #pragma unroll
    for (int i = 0; i < 32; i += 8)
        WT[(size_t)(n0 + ty + i) * EMBED + k0 + tx] = (bf16_t)tile[tx][ty + i];
}

// ---------------------------------------------------------------- GEMM core
// WG = 256 thr (8 waves). Tile 128(M) x 64(N), K-step 32, double-buffered LDS
// fed by async global->LDS copies. Each wave owns a 16x64 slice (4 f32x8 acc).

__device__ __forceinline__ void stage_tiles(
    const bf16_t* __restrict__ X, const bf16_t* __restrict__ WT,
    int m0, int n0, int kk, bf16_t* sA, bf16_t* sB)
{
    const int tid = threadIdx.x;
    const int row = tid >> 1, ch = tid & 1;
    const bf16_t* gA = X + (size_t)(m0 + row) * EMBED + kk + ch * 16;
    unsigned int la = (unsigned int)(uintptr_t)(sA + row * 48 + ch * 16);
    async_copy16(la,      gA);
    async_copy16(la + 16, gA + 8);
    if (tid < 128) {   // waves 0..3 stage B: 64 rows (n) x 32 k
        const bf16_t* gB = WT + (size_t)(n0 + row) * EMBED + kk + ch * 16;
        unsigned int lb = (unsigned int)(uintptr_t)(sB + row * 48 + ch * 16);
        async_copy16(lb,      gB);
        async_copy16(lb + 16, gB + 8);
    }
}

__device__ __forceinline__ void gemm_mainloop(
    const bf16_t* __restrict__ X, const bf16_t* __restrict__ WT,
    int m0, int n0, bf16_t* sA0, bf16_t* sB0, bf16_t* sA1, bf16_t* sB1,
    f32x8 acc[4])
{
    const int tid  = threadIdx.x;
    const int w    = tid >> 5;
    const int lane = tid & 31;
    const int r    = lane & 15;
    const int hf   = lane >> 4;

    stage_tiles(X, WT, m0, n0, 0, sA0, sB0);

    int cur = 0;
    for (int kk = 0; kk < EMBED; kk += 32, cur ^= 1) {
        wait_async0();        // this thread's copies into buffer `cur` are done
        __syncthreads();      // everyone's copies done; prev readers done too

        bf16_t* sA = cur ? sA1 : sA0;
        bf16_t* sB = cur ? sB1 : sB0;
        if (kk + 32 < EMBED)  // prefetch next tile into the other buffer
            stage_tiles(X, WT, m0, n0, kk + 32, cur ? sA0 : sA1, cur ? sB0 : sB1);

        // A fragment: 16x32, lane r = row, elems 0-15 -> K=hf*16..hf*16+15
        bf16x16 a;
        const bf16_t* pa = sA + (w * 16 + r) * 48;
        ((bf16x8*)&a)[0] = *(const bf16x8*)(pa + hf * 16);
        ((bf16x8*)&a)[1] = *(const bf16x8*)(pa + hf * 16 + 8);

        // hoist all 4 B fragments so DS loads pipeline ahead of the WMMA chain
        bf16x16 bfr[4];
        #pragma unroll
        for (int j = 0; j < 4; j++) {
            const bf16_t* pb = sB + (j * 16 + r) * 48 + hf * 16;
            ((bf16x8*)&bfr[j])[0] = *(const bf16x8*)(pb);
            ((bf16x8*)&bfr[j])[1] = *(const bf16x8*)(pb + 8);
        }
        #pragma unroll
        for (int j = 0; j < 4; j++)
            acc[j] = __builtin_amdgcn_wmma_f32_16x16x32_bf16(
                false, a, false, bfr[j], (short)0, acc[j], false, false);
    }
}

// mode 0: Q (scaled) -> [B,H,T,64] ; 1: K -> [B,H,T,64] ; 2: V -> [B,H,64,T]
__global__ __launch_bounds__(256)
void qkv_gemm(const bf16_t* __restrict__ X, const bf16_t* __restrict__ WT,
              const float* __restrict__ bias, bf16_t* __restrict__ out, int mode)
{
    __shared__ __align__(16) bf16_t sA0[128 * 48];
    __shared__ __align__(16) bf16_t sB0[64 * 48];
    __shared__ __align__(16) bf16_t sA1[128 * 48];
    __shared__ __align__(16) bf16_t sB1[64 * 48];
    const int m0 = blockIdx.x * 128, n0 = blockIdx.y * 64;
    const int w = threadIdx.x >> 5, lane = threadIdx.x & 31;
    const int r = lane & 15, hf = lane >> 4;

    f32x8 acc[4] = {};
    gemm_mainloop(X, WT, m0, n0, sA0, sB0, sA1, sB1, acc);

    #pragma unroll
    for (int j = 0; j < 4; j++) {
        int n = n0 + j * 16 + r;
        int h = n >> 6, d = n & 63;
        float bs = bias[n];
        #pragma unroll
        for (int v = 0; v < 8; v++) {
            int m = m0 + w * 16 + (hf << 3) + v;      // C layout: row = vgpr (+8 hi half)
            int bb = m >> 10, t = m & 1023;
            float val = acc[j][v] + bs;
            if (mode == 0)
                out[((size_t)(bb * NHEAD + h) * SEQ + t) * HDIM + d] = (bf16_t)(val * QK_SCALE);
            else if (mode == 1)
                out[((size_t)(bb * NHEAD + h) * SEQ + t) * HDIM + d] = (bf16_t)val;
            else
                out[((size_t)(bb * NHEAD + h) * HDIM + d) * SEQ + t] = (bf16_t)val;
        }
    }
}

__global__ __launch_bounds__(256)
void out_proj_gemm(const bf16_t* __restrict__ Ain, const bf16_t* __restrict__ WT,
                   const float* __restrict__ bias, float* __restrict__ out)
{
    __shared__ __align__(16) bf16_t sA0[128 * 48];
    __shared__ __align__(16) bf16_t sB0[64 * 48];
    __shared__ __align__(16) bf16_t sA1[128 * 48];
    __shared__ __align__(16) bf16_t sB1[64 * 48];
    const int m0 = blockIdx.x * 128, n0 = blockIdx.y * 64;
    const int w = threadIdx.x >> 5, lane = threadIdx.x & 31;
    const int r = lane & 15, hf = lane >> 4;

    f32x8 acc[4] = {};
    gemm_mainloop(Ain, WT, m0, n0, sA0, sB0, sA1, sB1, acc);

    #pragma unroll
    for (int j = 0; j < 4; j++) {
        int n = n0 + j * 16 + r;
        float bs = bias[n];
        #pragma unroll
        for (int v = 0; v < 8; v++) {
            int m = m0 + w * 16 + (hf << 3) + v;
            out[(size_t)m * EMBED + n] = acc[j][v] + bs;
        }
    }
}

// ---------------------------------------------------------------- flash attention
// Q,K: bf16 [B,H,T,64] (Q pre-scaled); VT: bf16 [B,H,64,T]; out: bf16 [B,T,E].
// One wave = 16 query rows, WG = 8 waves = 128 rows. grid = (T/128, B*H).

__global__ __launch_bounds__(256)
void flash_attn(const bf16_t* __restrict__ Q, const bf16_t* __restrict__ K,
                const bf16_t* __restrict__ VT, bf16_t* __restrict__ Aout)
{
    __shared__ __align__(16) bf16_t sP[8][16 * 32];
    const int w = threadIdx.x >> 5, lane = threadIdx.x & 31;
    const int r = lane & 15, hf = lane >> 4;
    const int bh = blockIdx.y;
    const int q0 = blockIdx.x * 128 + w * 16;

    const bf16_t* Qb = Q  + (size_t)bh * SEQ * HDIM;
    const bf16_t* Kb = K  + (size_t)bh * SEQ * HDIM;
    const bf16_t* Vb = VT + (size_t)bh * HDIM * SEQ;

    // Q fragments (d=0..31, d=32..63), A-layout
    bf16x16 aq0, aq1;
    {
        const bf16_t* p = Qb + (size_t)(q0 + r) * HDIM;
        ((bf16x8*)&aq0)[0] = *(const bf16x8*)(p + hf * 16);
        ((bf16x8*)&aq0)[1] = *(const bf16x8*)(p + hf * 16 + 8);
        ((bf16x8*)&aq1)[0] = *(const bf16x8*)(p + 32 + hf * 16);
        ((bf16x8*)&aq1)[1] = *(const bf16x8*)(p + 40 + hf * 16);
    }

    f32x8 acc0 = {}, acc1 = {}, acc2 = {}, acc3 = {};
    float mrow[8], lrow[8];
    #pragma unroll
    for (int v = 0; v < 8; v++) { mrow[v] = -1e30f; lrow[v] = 0.f; }

    bf16_t* P = sP[w];

    for (int kb = 0; kb <= q0 + 15; kb += 32) {
        // S = Q . K^T for 32 keys (two 16-wide N tiles)
        f32x8 S[2];
        #pragma unroll
        for (int jj = 0; jj < 2; jj++) {
            const bf16_t* pk = Kb + (size_t)(kb + jj * 16 + r) * HDIM + hf * 16;
            bf16x16 b0, b1;
            ((bf16x8*)&b0)[0] = *(const bf16x8*)(pk);
            ((bf16x8*)&b0)[1] = *(const bf16x8*)(pk + 8);
            ((bf16x8*)&b1)[0] = *(const bf16x8*)(pk + 32);
            ((bf16x8*)&b1)[1] = *(const bf16x8*)(pk + 40);
            f32x8 z = {};
            z = __builtin_amdgcn_wmma_f32_16x16x32_bf16(false, aq0, false, b0, (short)0, z, false, false);
            z = __builtin_amdgcn_wmma_f32_16x16x32_bf16(false, aq1, false, b1, (short)0, z, false, false);
            S[jj] = z;
        }

        const int kcol0 = kb + r, kcol1 = kb + 16 + r;
        const int qbase = q0 + (hf << 3);

        // online softmax per row (row = vgpr index within half)
        #pragma unroll
        for (int v = 0; v < 8; v++) {
            int q = qbase + v;
            float s0 = (kcol0 <= q) ? S[0][v] : -1e30f;
            float s1 = (kcol1 <= q) ? S[1][v] : -1e30f;
            float mx = fmaxf(s0, s1);
            #pragma unroll
            for (int off = 8; off; off >>= 1) mx = fmaxf(mx, __shfl_xor(mx, off, 32));
            float mnew  = fmaxf(mrow[v], mx);
            float resc  = __expf(mrow[v] - mnew);
            float p0    = __expf(s0 - mnew);
            float p1    = __expf(s1 - mnew);
            float rs = p0 + p1;
            #pragma unroll
            for (int off = 8; off; off >>= 1) rs += __shfl_xor(rs, off, 32);
            lrow[v] = lrow[v] * resc + rs;
            mrow[v] = mnew;
            acc0[v] = acc0[v] * resc;
            acc1[v] = acc1[v] * resc;
            acc2[v] = acc2[v] * resc;
            acc3[v] = acc3[v] * resc;
            int prow = (hf << 3) + v;
            P[prow * 32 + r]      = (bf16_t)p0;
            P[prow * 32 + 16 + r] = (bf16_t)p1;
        }
        asm volatile("s_wait_dscnt 0x0" ::: "memory");   // same-wave LDS RAW fence

        // P (C-layout -> A-layout via LDS)
        bf16x16 ap;
        {
            const bf16_t* pp = P + r * 32;
            ((bf16x8*)&ap)[0] = *(const bf16x8*)(pp + hf * 16);
            ((bf16x8*)&ap)[1] = *(const bf16x8*)(pp + hf * 16 + 8);
        }
        // O += P . V  (V^T storage -> contiguous B fragments)
        #pragma unroll
        for (int j = 0; j < 4; j++) {
            const bf16_t* pv = Vb + (size_t)(j * 16 + r) * SEQ + kb + hf * 16;
            bf16x16 bv;
            ((bf16x8*)&bv)[0] = *(const bf16x8*)(pv);
            ((bf16x8*)&bv)[1] = *(const bf16x8*)(pv + 8);
            f32x8& A = (j == 0 ? acc0 : j == 1 ? acc1 : j == 2 ? acc2 : acc3);
            A = __builtin_amdgcn_wmma_f32_16x16x32_bf16(false, ap, false, bv, (short)0, A, false, false);
        }
    }

    // normalize + store bf16 [B,T,E]
    const int bIdx = bh / NHEAD, h = bh % NHEAD;
    #pragma unroll
    for (int v = 0; v < 8; v++) {
        float inv = 1.0f / lrow[v];
        int q = q0 + (hf << 3) + v;
        bf16_t* o = Aout + ((size_t)(bIdx * SEQ + q)) * EMBED + h * HDIM + r;
        o[0]  = (bf16_t)(acc0[v] * inv);
        o[16] = (bf16_t)(acc1[v] * inv);
        o[32] = (bf16_t)(acc2[v] * inv);
        o[48] = (bf16_t)(acc3[v] * inv);
    }
}

// ---------------------------------------------------------------- launcher

extern "C" void kernel_launch(void* const* d_in, const int* in_sizes, int n_in,
                              void* d_out, int out_size, void* d_ws, size_t ws_size,
                              hipStream_t stream)
{
    const float* x  = (const float*)d_in[0];
    const float* Wq = (const float*)d_in[1];
    const float* bq = (const float*)d_in[2];
    const float* Wk = (const float*)d_in[3];
    const float* bk = (const float*)d_in[4];
    const float* Wv = (const float*)d_in[5];
    const float* bv = (const float*)d_in[6];
    const float* Wo = (const float*)d_in[7];
    const float* bo = (const float*)d_in[8];
    float* out = (float*)d_out;

    char* ws = (char*)d_ws;
    const size_t SZ_X = (size_t)NBATCH * SEQ * EMBED * sizeof(bf16_t);   // 12.58 MB
    const size_t SZ_W = (size_t)EMBED * EMBED * sizeof(bf16_t);          // 1.18 MB
    bf16_t* xb  = (bf16_t*)(ws);
    bf16_t* wqt = (bf16_t*)(ws + SZ_X);
    bf16_t* wkt = (bf16_t*)(ws + SZ_X + SZ_W);
    bf16_t* wvt = (bf16_t*)(ws + SZ_X + 2 * SZ_W);
    bf16_t* wot = (bf16_t*)(ws + SZ_X + 3 * SZ_W);
    bf16_t* qb  = (bf16_t*)(ws + SZ_X + 4 * SZ_W);
    bf16_t* kbuf= (bf16_t*)(ws + 2 * SZ_X + 4 * SZ_W);
    bf16_t* vtb = (bf16_t*)(ws + 3 * SZ_X + 4 * SZ_W);
    bf16_t* ab  = (bf16_t*)(ws + 4 * SZ_X + 4 * SZ_W);

    // 1) convert x to bf16
    {
        int n4 = NBATCH * SEQ * EMBED / 4;
        cvt_f32_bf16<<<(n4 + 255) / 256, 256, 0, stream>>>(x, xb, n4);
    }
    // 2) transpose + convert weights
    {
        dim3 g(EMBED / 32, EMBED / 32), b(32, 8);
        transpose_w_bf16<<<g, b, 0, stream>>>(Wq, wqt);
        transpose_w_bf16<<<g, b, 0, stream>>>(Wk, wkt);
        transpose_w_bf16<<<g, b, 0, stream>>>(Wv, wvt);
        transpose_w_bf16<<<g, b, 0, stream>>>(Wo, wot);
    }
    // 3) QKV projections
    {
        dim3 g(NBATCH * SEQ / 128, EMBED / 64);
        qkv_gemm<<<g, 256, 0, stream>>>(xb, wqt, bq, qb,  0);
        qkv_gemm<<<g, 256, 0, stream>>>(xb, wkt, bk, kbuf, 1);
        qkv_gemm<<<g, 256, 0, stream>>>(xb, wvt, bv, vtb, 2);
    }
    // 4) causal flash attention
    {
        dim3 g(SEQ / 128, NBATCH * NHEAD);
        flash_attn<<<g, 256, 0, stream>>>(qb, kbuf, vtb, ab);
    }
    // 5) output projection -> fp32
    {
        dim3 g(NBATCH * SEQ / 128, EMBED / 64);
        out_proj_gemm<<<g, 256, 0, stream>>>(ab, wot, bo, out);
    }
}
